// MambaMLPLayer_46119358824485
// MI455X (gfx1250) — matrix operand order repaired and checked by reading, not compiled, not run on hardware
//
#include <hip/hip_runtime.h>
#include <hip/hip_bf16.h>

// ---------------------------------------------------------------------------
// MI455X (gfx1250) Mamba+MLP layer.
// GEMMs via v_wmma_f32_16x16x32_f16 (wave32, 16x16x32 fragments).
// Block tile 128x128, 8 waves (4M x 2N), wave tile 32x64, BK=64:
// 16 WMMAs per LDS staging phase.
// ---------------------------------------------------------------------------

typedef _Float16 h8   __attribute__((ext_vector_type(8)));
typedef _Float16 v16h __attribute__((ext_vector_type(16)));
typedef float    v8f  __attribute__((ext_vector_type(8)));

#define BB 2
#define LL 4096
#define DM 512
#define DI 2048
#define DS 64
#define RK 32
#define NROWS (BB * LL)     // 8192

// ---- GEMM tile config ----
#define BM 128
#define BN 128
#define BK 64
#define LDSB 72             // padded row stride in halves (144B, 16B aligned)

enum EpMode {
    EP_F32 = 0,             // out_f32 = v
    EP_SOFTPLUS_BIAS = 1,   // out_f32 = softplus(v + bias[n])
    EP_RESIDUAL = 2,        // out_f32 = v + res[m*N+n]
    EP_BIAS_GELU_F16 = 3,   // out_f16 = gelu_exact(v + bias[n])
    EP_BIAS_RESIDUAL = 4,   // out_f32 = v + bias[n] + res[m*N+n]
};

// OUT[M,N] = epilogue( A16[M,K] @ W16[N,K]^T )
// M multiple of 128, K multiple of 32, N arbitrary (guarded).
__global__ __launch_bounds__(256)
void wmma_gemm_kernel(const _Float16* __restrict__ A,
                      const _Float16* __restrict__ W,
                      int M, int N, int K,
                      float* __restrict__ outf32,
                      _Float16* __restrict__ outf16,
                      const float* __restrict__ bias,
                      const float* __restrict__ residual,
                      int mode)
{
    __shared__ _Float16 As[BM * LDSB];   // 18 KB
    __shared__ _Float16 Bs[BN * LDSB];   // 18 KB, stored [N][K]

    const int tid  = threadIdx.x;
    const int lane = tid & 31;
    const int wave = tid >> 5;          // 0..7
    const int wm   = wave & 3;          // 4 M-groups of 32
    const int wn   = wave >> 2;         // 2 N-groups of 64

    const int m0 = blockIdx.y * BM;
    const int n0 = blockIdx.x * BN;

    v8f acc[2][4] = {};

    const int mrow = lane & 15;
    const int ak   = (lane < 16) ? 0 : 8;    // A frag K base (interleaved layout)
    const int bk   = (lane < 16) ? 0 : 8;    // B frag K base (interleaved layout)

    // staging map: each thread loads one 32-half half-row of A and of B
    const int srow = tid >> 1;           // 0..127
    const int scol = (tid & 1) * 32;     // 0 or 32

    for (int k0 = 0; k0 < K; k0 += BK) {
        // ---- stage A tile: 128x64 halves (zero-fill beyond K) ----
        {
            const _Float16* gp = A + (size_t)(m0 + srow) * K + k0 + scol;
            _Float16* lp = As + srow * LDSB + scol;
#pragma unroll
            for (int j = 0; j < 4; ++j) {
                h8 v = {};
                if (k0 + scol + j * 8 < K) v = *(const h8*)(gp + j * 8);
                *(h8*)(lp + j * 8) = v;
            }
            if (k0 + BK < K) __builtin_prefetch(gp + BK, 0, 1);
        }
        // ---- stage W tile as B[N][K]: 128x64 halves ----
        {
            const int gn = n0 + srow;
            const _Float16* gp = W + (size_t)gn * K + k0 + scol;
            _Float16* lp = Bs + srow * LDSB + scol;
#pragma unroll
            for (int j = 0; j < 4; ++j) {
                h8 v = {};
                if (gn < N && (k0 + scol + j * 8 < K)) v = *(const h8*)(gp + j * 8);
                *(h8*)(lp + j * 8) = v;
            }
            if (gn < N && (k0 + BK < K)) __builtin_prefetch(gp + BK, 0, 1);
        }
        __syncthreads();

        // ---- two k-steps of 32 over the staged 64-wide tile ----
#pragma unroll
        for (int ks = 0; ks < 2; ++ks) {
            v16h afrag[2], bfrag[4];
#pragma unroll
            for (int mi = 0; mi < 2; ++mi) {
                const _Float16* base =
                    As + (wm * 32 + mi * 16 + mrow) * LDSB + ks * 32 + ak;
                h8 lo = *(const h8*)(base);
                h8 hi = *(const h8*)(base + 16);
                afrag[mi] = __builtin_shufflevector(lo, hi,
                    0,1,2,3,4,5,6,7,8,9,10,11,12,13,14,15);
            }
#pragma unroll
            for (int ni = 0; ni < 4; ++ni) {
                const _Float16* base =
                    Bs + (wn * 64 + ni * 16 + mrow) * LDSB + ks * 32 + bk;
                h8 lo = *(const h8*)(base);
                h8 hi = *(const h8*)(base + 16);
                bfrag[ni] = __builtin_shufflevector(lo, hi,
                    0,1,2,3,4,5,6,7,8,9,10,11,12,13,14,15);
            }
#pragma unroll
            for (int mi = 0; mi < 2; ++mi)
#pragma unroll
                for (int ni = 0; ni < 4; ++ni)
                    acc[mi][ni] = __builtin_amdgcn_wmma_f32_16x16x32_f16(
                        false, afrag[mi], false, bfrag[ni],
                        (short)0, acc[mi][ni], false, false);
        }
        __syncthreads();
    }

    // ---- epilogue: C layout = lane(0-15): M=r,N=lane ; lane(16-31): M=8+r ----
    const int nbase = n0 + wn * 64;
    const int mbase = m0 + wm * 32;
#pragma unroll
    for (int mi = 0; mi < 2; ++mi) {
#pragma unroll
        for (int ni = 0; ni < 4; ++ni) {
            const int ncol  = nbase + ni * 16 + (lane & 15);
            const int mrow0 = mbase + mi * 16 + ((lane < 16) ? 0 : 8);
            if (ncol >= N) continue;
#pragma unroll
            for (int r = 0; r < 8; ++r) {
                const int m = mrow0 + r;
                float v = acc[mi][ni][r];
                const size_t off = (size_t)m * N + ncol;
                if (mode == EP_F32) {
                    outf32[off] = v;
                } else if (mode == EP_SOFTPLUS_BIAS) {
                    float t = v + bias[ncol];
                    outf32[off] = (t > 20.0f) ? t : log1pf(__expf(t));
                } else if (mode == EP_RESIDUAL) {
                    outf32[off] = v + residual[off];
                } else if (mode == EP_BIAS_GELU_F16) {
                    float t = v + bias[ncol];
                    outf16[off] = (_Float16)(0.5f * t * (1.0f + erff(t * 0.70710678f)));
                } else { // EP_BIAS_RESIDUAL
                    outf32[off] = v + bias[ncol] + residual[off];
                }
            }
        }
    }
}

// ---- LayerNorm (cols = 512) -> f16 output ----
__global__ __launch_bounds__(256)
void ln_f16_kernel(const float* __restrict__ x,
                   const float* __restrict__ g,
                   const float* __restrict__ b,
                   _Float16* __restrict__ out)
{
    __shared__ float red[256];
    const int row = blockIdx.x;
    const int t = threadIdx.x;
    const float* xr = x + (size_t)row * DM;
    float v0 = xr[t], v1 = xr[t + 256];

    red[t] = v0 + v1;
    __syncthreads();
    for (int w = 128; w > 0; w >>= 1) {
        if (t < w) red[t] += red[t + w];
        __syncthreads();
    }
    const float mean = red[0] * (1.0f / DM);
    __syncthreads();

    const float d0 = v0 - mean, d1 = v1 - mean;
    red[t] = d0 * d0 + d1 * d1;
    __syncthreads();
    for (int w = 128; w > 0; w >>= 1) {
        if (t < w) red[t] += red[t + w];
        __syncthreads();
    }
    const float inv = rsqrtf(red[0] * (1.0f / DM) + 1e-6f);

    _Float16* o = out + (size_t)row * DM;
    o[t]       = (_Float16)(d0 * inv * g[t]       + b[t]);
    o[t + 256] = (_Float16)(d1 * inv * g[t + 256] + b[t + 256]);
}

// ---- depthwise causal conv (K=4) + SiLU ; u read from xz[:, :DI] ----
__global__ __launch_bounds__(256)
void conv_silu_kernel(const float* __restrict__ xz,
                      const float* __restrict__ conv_w,
                      const float* __restrict__ conv_b,
                      _Float16* __restrict__ u16)
{
    const int idx = blockIdx.x * 256 + threadIdx.x;      // over B*L*DI
    const int d  = idx & (DI - 1);
    const int bl = idx >> 11;                            // b*L + l
    const int l  = bl & (LL - 1);
    float acc = conv_b[d];
    const float* w = conv_w + d * 4;
    const size_t rowbase = (size_t)(bl - l) * (2 * DI);  // b*L rows
#pragma unroll
    for (int k = 0; k < 4; ++k) {
        const int lp = l - 3 + k;
        if (lp >= 0)
            acc += w[k] * xz[rowbase + (size_t)lp * (2 * DI) + d];
    }
    const float sig = 1.0f / (1.0f + __expf(-acc));
    u16[(size_t)bl * DI + d] = (_Float16)(acc * sig);
}

// ---- selective scan: one channel (b,d) per lane; h[64] in VGPRs ----
// Fuses +u*D and *silu(z); writes gated y in place over u16.
__global__ __launch_bounds__(128)
void scan_kernel(const float* __restrict__ delta,
                 _Float16* __restrict__ u16,        // in: u  out: y*silu(z)
                 const float* __restrict__ xz,      // z = xz[:, DI:2*DI]
                 const float* __restrict__ xdbl,    // B,C = xdbl[:, RK : RK+2*DS]
                 const float* __restrict__ A_log,
                 const float* __restrict__ Dvec)
{
    __shared__ float Ash[DS * 128];   // 32 KB, [s][tid] -> conflict free
    __shared__ float BC[2 * DS];      // B_t then C_t

    const int t  = threadIdx.x;
    const int ch = blockIdx.x * 128 + t;   // 0 .. B*DI-1 (same b per block)
    const int b  = ch >> 11;
    const int d  = ch & (DI - 1);

    for (int s = 0; s < DS; ++s)
        Ash[s * 128 + t] = -__expf(A_log[(size_t)d * DS + s]);
    const float Dd = Dvec[d];

    float h[DS];
#pragma unroll
    for (int s = 0; s < DS; ++s) h[s] = 0.0f;
    __syncthreads();

    for (int l = 0; l < LL; ++l) {
        const size_t bl = (size_t)(b * LL + l);
        if (t < 2 * DS) BC[t] = xdbl[bl * (RK + 2 * DS) + RK + t];
        __syncthreads();

        const float dt = delta[bl * DI + d];
        const float u  = (float)u16[bl * DI + d];
        const float z  = xz[bl * (2 * DI) + DI + d];
        const float du = dt * u;
        float y = 0.0f;
#pragma unroll
        for (int s = 0; s < DS; ++s) {
            const float dA = __expf(dt * Ash[s * 128 + t]);
            const float hn = h[s] * dA + du * BC[s];
            h[s] = hn;
            y += hn * BC[DS + s];
        }
        y += u * Dd;
        const float sig = 1.0f / (1.0f + __expf(-z));
        u16[bl * DI + d] = (_Float16)(y * (z * sig));
        __syncthreads();
    }
}

// ---- strided f32 -> f16 convert ----
__global__ __launch_bounds__(256)
void cvt_f32_f16_kernel(const float* __restrict__ src, _Float16* __restrict__ dst,
                        int rows, int cols, int ld)
{
    const int idx = blockIdx.x * 256 + threadIdx.x;
    if (idx >= rows * cols) return;
    const int r = idx / cols;
    const int c = idx - r * cols;
    dst[idx] = (_Float16)src[(size_t)r * ld + c];
}

// ---------------------------------------------------------------------------
extern "C" void kernel_launch(void* const* d_in, const int* in_sizes, int n_in,
                              void* d_out, int out_size, void* d_ws, size_t ws_size,
                              hipStream_t stream)
{
    const float* x          = (const float*)d_in[0];
    const float* ln1_g      = (const float*)d_in[1];
    const float* ln1_b      = (const float*)d_in[2];
    const float* in_proj_w  = (const float*)d_in[3];
    const float* conv_w     = (const float*)d_in[4];
    const float* conv_b     = (const float*)d_in[5];
    const float* x_proj_w   = (const float*)d_in[6];
    const float* dt_proj_w  = (const float*)d_in[7];
    const float* dt_proj_b  = (const float*)d_in[8];
    const float* A_log      = (const float*)d_in[9];
    const float* Dvec       = (const float*)d_in[10];
    const float* out_proj_w = (const float*)d_in[11];
    const float* ln2_g      = (const float*)d_in[12];
    const float* ln2_b      = (const float*)d_in[13];
    const float* mlp_w1     = (const float*)d_in[14];
    const float* mlp_b1     = (const float*)d_in[15];
    const float* mlp_w2     = (const float*)d_in[16];
    const float* mlp_b2     = (const float*)d_in[17];
    float* out = (float*)d_out;

    // ---- workspace carve ----
    size_t off = 0;
    auto carve = [&](size_t bytes) -> void* {
        off = (off + 255) & ~(size_t)255;
        void* p = (char*)d_ws + off;
        off += bytes;
        return p;
    };
    _Float16* w16_inproj  = (_Float16*)carve((size_t)(2*DI) * DM * 2);
    _Float16* w16_xproj   = (_Float16*)carve((size_t)(RK + 2*DS) * DI * 2);
    _Float16* w16_dtproj  = (_Float16*)carve((size_t)DI * RK * 2);
    _Float16* w16_outproj = (_Float16*)carve((size_t)DM * DI * 2);
    _Float16* w16_mlp1    = (_Float16*)carve((size_t)(4*DM) * DM * 2);
    _Float16* w16_mlp2    = (_Float16*)carve((size_t)DM * (4*DM) * 2);
    _Float16* ln16        = (_Float16*)carve((size_t)NROWS * DM * 2);
    float*    xz          = (float*)   carve((size_t)NROWS * (2*DI) * 4);
    _Float16* u16         = (_Float16*)carve((size_t)NROWS * DI * 2);
    float*    xdbl        = (float*)   carve((size_t)NROWS * (RK + 2*DS) * 4);
    _Float16* dt16        = (_Float16*)carve((size_t)NROWS * RK * 2);
    float*    deltab      = (float*)   carve((size_t)NROWS * DI * 4);
    float*    x2          = (float*)   carve((size_t)NROWS * DM * 4);
    _Float16* hid16       = (_Float16*)deltab;   // reuse (delta dead after scan)
    (void)in_sizes; (void)n_in; (void)out_size; (void)ws_size;

    auto cvt = [&](const float* s, _Float16* dst, int rows, int cols) {
        int n = rows * cols;
        cvt_f32_f16_kernel<<<(n + 255) / 256, 256, 0, stream>>>(s, dst, rows, cols, cols);
    };
    // weight conversions (f32 -> f16)
    cvt(in_proj_w,  w16_inproj,  2*DI, DM);
    cvt(x_proj_w,   w16_xproj,   RK + 2*DS, DI);
    cvt(dt_proj_w,  w16_dtproj,  DI, RK);
    cvt(out_proj_w, w16_outproj, DM, DI);
    cvt(mlp_w1,     w16_mlp1,    4*DM, DM);
    cvt(mlp_w2,     w16_mlp2,    DM, 4*DM);

    auto gemm = [&](const _Float16* A, const _Float16* W, int M, int N, int K,
                    float* of32, _Float16* of16, const float* bias,
                    const float* res, int mode) {
        dim3 grid((N + BN - 1) / BN, M / BM);
        wmma_gemm_kernel<<<grid, 256, 0, stream>>>(A, W, M, N, K, of32, of16,
                                                   bias, res, mode);
    };

    // 1) LN1 -> f16
    ln_f16_kernel<<<NROWS, 256, 0, stream>>>(x, ln1_g, ln1_b, ln16);
    // 2) in_proj: xz = ln @ Wip^T   [8192 x 4096]
    gemm(ln16, w16_inproj, NROWS, 2*DI, DM, xz, nullptr, nullptr, nullptr, EP_F32);
    // 3) depthwise conv + SiLU -> u16
    conv_silu_kernel<<<(NROWS * DI) / 256, 256, 0, stream>>>(xz, conv_w, conv_b, u16);
    // 4) x_proj: xdbl = u @ Wxp^T   [8192 x 160]
    gemm(u16, w16_xproj, NROWS, RK + 2*DS, DI, xdbl, nullptr, nullptr, nullptr, EP_F32);
    // 5) dt slice -> f16
    cvt_f32_f16_kernel<<<(NROWS * RK + 255) / 256, 256, 0, stream>>>(
        xdbl, dt16, NROWS, RK, RK + 2*DS);
    // 6) dt_proj + softplus(.+b): delta   [8192 x 2048]
    gemm(dt16, w16_dtproj, NROWS, DI, RK, deltab, nullptr, dt_proj_b, nullptr,
         EP_SOFTPLUS_BIAS);
    // 7) selective scan (fused u*D + silu(z) gate), in-place over u16
    scan_kernel<<<(BB * DI) / 128, 128, 0, stream>>>(deltab, u16, xz, xdbl,
                                                     A_log, Dvec);
    // 8) out_proj + residual(x): x2   [8192 x 512]
    gemm(u16, w16_outproj, NROWS, DM, DI, x2, nullptr, nullptr, x, EP_RESIDUAL);
    // 9) LN2 -> f16
    ln_f16_kernel<<<NROWS, 256, 0, stream>>>(x2, ln2_g, ln2_b, ln16);
    // 10) MLP up + bias + exact GELU -> f16 hidden  [8192 x 2048]
    gemm(ln16, w16_mlp1, NROWS, 4*DM, DM, nullptr, hid16, mlp_b1, nullptr,
         EP_BIAS_GELU_F16);
    // 11) MLP down + bias + residual(x2) -> out  [8192 x 512]
    gemm(hid16, w16_mlp2, NROWS, DM, 4*DM, out, nullptr, mlp_b2, x2,
         EP_BIAS_RESIDUAL);
}
